// DGCNN_Grouper_14293651161199
// MI455X (gfx1250) — matrix operand order repaired and checked by reading, not compiled, not run on hardware
//
#include <hip/hip_runtime.h>
#include <hip/hip_bf16.h>

typedef __attribute__((ext_vector_type(16))) _Float16 v16h;
typedef __attribute__((ext_vector_type(8)))  float    v8f;

#define NBATCH 16
#define KNN_K  16
#define LEAK   0.2f
#define GN_EPS 1e-5f

// one-instruction XOR butterfly within each 16-lane half (ds_swizzle group-of-32:
// offset = xor_mask<<10 | and_mask(0x1f); xor<16 stays inside the half-wave)
template <int MASK>
__device__ __forceinline__ float swz_xor(float v) {
    return __int_as_float(
        __builtin_amdgcn_ds_swizzle(__float_as_int(v), (MASK << 10) | 0x1F));
}

// ---------------------------------------------------------------- utilities
__global__ void k_zero(float* p, int n) {
    int t = blockIdx.x * blockDim.x + threadIdx.x;
    if (t < n) p[t] = 0.f;
}

__global__ void k_cvtw(const float* __restrict__ w, _Float16* __restrict__ wh, int n) {
    int t = blockIdx.x * blockDim.x + threadIdx.x;
    if (t < n) wh[t] = (_Float16)w[t];
}

// ------------------------------------------------- input projection (8x3 GEMV)
// coor: (B,3,N) ; f0: (B,N,8)  [features are N-major everywhere]
__global__ void k_input(const float* __restrict__ x, const float* __restrict__ w_in,
                        const float* __restrict__ b_in, float* __restrict__ coor,
                        float* __restrict__ f0) {
    int t = blockIdx.x * blockDim.x + threadIdx.x;
    if (t >= NBATCH * 2048) return;
    int b = t / 2048, n = t % 2048;
    float p[3];
#pragma unroll
    for (int d = 0; d < 3; ++d) {
        p[d] = x[((size_t)b * 2048 + n) * 3 + d];
        coor[((size_t)b * 3 + d) * 2048 + n] = p[d];
    }
#pragma unroll
    for (int o = 0; o < 8; ++o) {
        float v = b_in[o];
#pragma unroll
        for (int d = 0; d < 3; ++d) v += w_in[o * 3 + d] * p[d];
        f0[((size_t)b * 2048 + n) * 8 + o] = v;
    }
}

// ---------------------------------------------------------------- KNN (top-16)
__global__ void k_knn(const float* __restrict__ cq, const float* __restrict__ ck,
                      int* __restrict__ idx, int Nq, int Nk, int bpb) {
    __shared__ float skx[2048], sky[2048], skz[2048];
    int b = blockIdx.x / bpb;
    int q = (blockIdx.x % bpb) * 256 + threadIdx.x;
    for (int j = threadIdx.x; j < Nk; j += 256) {
        skx[j] = ck[((size_t)b * 3 + 0) * Nk + j];
        sky[j] = ck[((size_t)b * 3 + 1) * Nk + j];
        skz[j] = ck[((size_t)b * 3 + 2) * Nk + j];
    }
    __syncthreads();
    if (q >= Nq) return;
    float qx = cq[((size_t)b * 3 + 0) * Nq + q];
    float qy = cq[((size_t)b * 3 + 1) * Nq + q];
    float qz = cq[((size_t)b * 3 + 2) * Nq + q];
    float bd[KNN_K]; int bi[KNN_K];
#pragma unroll
    for (int j = 0; j < KNN_K; ++j) { bd[j] = 3.0e38f; bi[j] = 0; }
    for (int jj = 0; jj < Nk; ++jj) {
        float dx = qx - skx[jj], dy = qy - sky[jj], dz = qz - skz[jj];
        float d = dx * dx + dy * dy + dz * dz;
        int ci = jj;
        if (d < bd[KNN_K - 1]) {
#pragma unroll
            for (int j = 0; j < KNN_K; ++j) {
                if (d < bd[j]) {
                    float td = bd[j]; int ti = bi[j];
                    bd[j] = d; bi[j] = ci; d = td; ci = ti;
                }
            }
        }
    }
#pragma unroll
    for (int j = 0; j < KNN_K; ++j) idx[((size_t)b * Nq + q) * KNN_K + j] = bi[j];
}

// ------------------------------------------------------------------------ FPS
__global__ void k_fps(const float* __restrict__ coor, int* __restrict__ fidx,
                      int N, int npoint) {
    __shared__ float sx[2048], sy[2048], sz[2048], sd[2048];
    __shared__ float rv[256];
    __shared__ int   ra[256];
    __shared__ int   sfar;
    int b = blockIdx.x;
    for (int j = threadIdx.x; j < N; j += 256) {
        sx[j] = coor[((size_t)b * 3 + 0) * N + j];
        sy[j] = coor[((size_t)b * 3 + 1) * N + j];
        sz[j] = coor[((size_t)b * 3 + 2) * N + j];
        sd[j] = 1.0e10f;
    }
    if (threadIdx.x == 0) sfar = 0;
    __syncthreads();
    for (int it = 0; it < npoint; ++it) {
        int far = sfar;
        if (threadIdx.x == 0) fidx[b * npoint + it] = far;
        float cx = sx[far], cy = sy[far], cz = sz[far];
        float lb = -1.f; int la = 0;
        for (int j = threadIdx.x; j < N; j += 256) {
            float dx = sx[j] - cx, dy = sy[j] - cy, dz = sz[j] - cz;
            float d = dx * dx + dy * dy + dz * dz;
            float nd = fminf(sd[j], d);
            sd[j] = nd;
            if (nd > lb) { lb = nd; la = j; }
        }
        rv[threadIdx.x] = lb; ra[threadIdx.x] = la;
        __syncthreads();
        for (int s = 128; s > 0; s >>= 1) {
            if (threadIdx.x < s) {
                float ov = rv[threadIdx.x + s]; int oa = ra[threadIdx.x + s];
                if (ov > rv[threadIdx.x] ||
                    (ov == rv[threadIdx.x] && oa < ra[threadIdx.x])) {
                    rv[threadIdx.x] = ov; ra[threadIdx.x] = oa;
                }
            }
            __syncthreads();
        }
        if (threadIdx.x == 0) sfar = ra[0];
        __syncthreads();
    }
}

// ------------------------------------------------------- FPS gather (coor + f)
__global__ void k_gather(const float* __restrict__ coor, const float* __restrict__ f,
                         const int* __restrict__ fidx, float* __restrict__ nc,
                         float* __restrict__ nf, int N, int C, int npoint) {
    int t = blockIdx.x * blockDim.x + threadIdx.x;
    if (t >= NBATCH * npoint) return;
    int b = t / npoint, i = t % npoint;
    int s = fidx[b * npoint + i];
#pragma unroll
    for (int d = 0; d < 3; ++d)
        nc[((size_t)b * 3 + d) * npoint + i] = coor[((size_t)b * 3 + d) * N + s];
    const float* src = f + ((size_t)b * N + s) * C;
    float* dst = nf + ((size_t)b * npoint + i) * C;
    for (int c = 0; c < C; ++c) dst[c] = src[c];
}

// --------------------------------------------- fused edge-GEMM (WMMA) + GN path
// Branch-free fragment build: for CK%16==0 the 16-channel run per (lane,ks) sits
// entirely in one region, so we select the base pointer once and issue 16
// unconditional contiguous loads (b128-coalescible) + fma; stage-1 (CK=8) uses
// compile-time per-h regions and one hi-select multiply.
template <int O, int CK, int MODE>
__global__ void k_gemm(const _Float16* __restrict__ Wh,
                       const float* __restrict__ fk, const float* __restrict__ fq,
                       const int* __restrict__ idx,
                       const float* __restrict__ gamma, const float* __restrict__ beta,
                       const float* __restrict__ mrs, float* __restrict__ stats,
                       float* __restrict__ fout, int Nq, int Nk) {
    constexpr int C2 = 2 * CK;
    constexpr int KS = (C2 + 31) / 32;

    const int lane = threadIdx.x & 31;
    const int wave = blockIdx.x * (blockDim.x >> 5) + (threadIdx.x >> 5);
    const int tilesPerB = (O >> 4) * Nq;
    const int b   = wave / tilesPerB;
    int rem       = wave - b * tilesPerB;
    const int oT  = rem / Nq;
    const int n   = rem - oT * Nq;
    const int o0  = oT << 4;
    const int hi  = lane >> 4;
    const int col = lane & 15;

    const int key = idx[((size_t)b * Nq + n) * KNN_K + col];
    const float* kbp = fk + ((size_t)b * Nk + key) * CK;   // neighbor channels
    const float* qbp = fq + ((size_t)b * Nq + n) * CK;     // query channels
    const _Float16* wrow = Wh + (size_t)(o0 + col) * C2;

    v8f acc = {0.f, 0.f, 0.f, 0.f, 0.f, 0.f, 0.f, 0.f};
#pragma unroll
    for (int ks = 0; ks < KS; ++ks) {
        v16h a, bm;
        if constexpr (CK == 8) {
            // stage 1: C2=16, KS=1. A: ka = h + hi*16, valid only for hi==0.
#pragma unroll
            for (int h = 0; h < 16; ++h)
                a[h] = hi ? (_Float16)0.0f : wrow[h];
            // B: lo half covers c=0..15 (0..7 diff, 8..15 xq); hi half -> zero
            const float vf = hi ? 0.0f : 1.0f;
            float kv[8], qv[8];
#pragma unroll
            for (int j = 0; j < 8; ++j) { kv[j] = kbp[j]; qv[j] = qbp[j]; }
#pragma unroll
            for (int h = 0; h < 8; ++h)  bm[h] = (_Float16)((kv[h] - qv[h]) * vf);
#pragma unroll
            for (int h = 8; h < 16; ++h) bm[h] = (_Float16)(qv[h - 8] * vf);
        } else {
            // A: ka = ks*32 + hi*16 + h -- always < C2 here
#pragma unroll
            for (int h = 0; h < 16; ++h)
                a[h] = wrow[ks * 32 + hi * 16 + h];
            // B: run [base, base+16) wholly in diff- or xq-region
            const int base = ks * 32 + hi * 16;
            const bool diff = base < CK;
            const float fs = diff ? 1.0f : 0.0f;
            const float* p1 = diff ? (kbp + base) : (qbp + (base - CK));
            const float* p2 = qbp + (diff ? base : (base - CK));
#pragma unroll
            for (int h = 0; h < 16; ++h)
                bm[h] = (_Float16)(p1[h] - fs * p2[h]);
        }
        acc = __builtin_amdgcn_wmma_f32_16x16x32_f16(false, a, false, bm,
                                                     (short)0, acc, false, false);
    }

    constexpr int cpg = O / 4;                    // channels per GN group
    const int g = (o0 + (hi ? 8 : 0)) / cpg;

    if (MODE == 0) {
        float s = 0.f, s2 = 0.f;
#pragma unroll
        for (int r = 0; r < 8; ++r) { float v = acc[r]; s += v; s2 += v * v; }
        s += swz_xor<1>(s);  s2 += swz_xor<1>(s2);
        s += swz_xor<2>(s);  s2 += swz_xor<2>(s2);
        s += swz_xor<4>(s);  s2 += swz_xor<4>(s2);
        s += swz_xor<8>(s);  s2 += swz_xor<8>(s2);
        if (col == 0) {
            atomicAdd(&stats[((b << 2) + g) * 2 + 0], s);
            atomicAdd(&stats[((b << 2) + g) * 2 + 1], s2);
        }
    } else {
        const float mu = mrs[((b << 2) + g) * 2 + 0];
        const float rs = mrs[((b << 2) + g) * 2 + 1];
#pragma unroll
        for (int r = 0; r < 8; ++r) {
            const int o = o0 + (hi ? 8 : 0) + r;          // D layout: M = r + hi*8
            float v = (acc[r] - mu) * rs * gamma[o] + beta[o];
            v = v > 0.f ? v : LEAK * v;
            v = fmaxf(v, swz_xor<1>(v));                  // max over 16 neighbors
            v = fmaxf(v, swz_xor<2>(v));
            v = fmaxf(v, swz_xor<4>(v));
            v = fmaxf(v, swz_xor<8>(v));
            if (col == 0) fout[((size_t)b * Nq + n) * O + o] = v;
        }
    }
}

__global__ void k_finalize(const float* __restrict__ stats, float* __restrict__ mrs,
                           float cnt) {
    int i = threadIdx.x;
    if (i < NBATCH * 4) {
        float s = stats[2 * i], s2 = stats[2 * i + 1];
        float mu = s / cnt;
        float var = s2 / cnt - mu * mu;
        mrs[2 * i]     = mu;
        mrs[2 * i + 1] = rsqrtf(var + GN_EPS);
    }
}

// ------------------------------------------------------------- final outputs
__global__ void k_out(const float* __restrict__ coor2, const float* __restrict__ f4,
                      float* __restrict__ out) {
    int t = blockIdx.x * blockDim.x + threadIdx.x;
    const int COORSZ = NBATCH * 128 * 3;
    const int FSZ = NBATCH * 128 * 128;
    if (t < COORSZ) {
        int b = t / (128 * 3), r = t % (128 * 3), n = r / 3, d = r % 3;
        out[t] = coor2[((size_t)b * 3 + d) * 128 + n];
    } else if (t < COORSZ + FSZ) {
        out[t] = f4[t - COORSZ];
    }
}

// ------------------------------------------------------------- staged driver
template <int O, int CK>
static void launch_stage(const float* w, const float* gamma, const float* beta,
                         const float* coorq, const float* coork,
                         const float* fq, const float* fk,
                         int Nq, int Nk, float* fout,
                         int* idx, _Float16* wh, float* stats, float* mrs,
                         hipStream_t stream) {
    constexpr int C2 = 2 * CK;
    int bpb = (Nq + 255) / 256;
    k_knn<<<NBATCH * bpb, 256, 0, stream>>>(coorq, coork, idx, Nq, Nk, bpb);
    k_cvtw<<<(O * C2 + 255) / 256, 256, 0, stream>>>(w, wh, O * C2);
    k_zero<<<1, 128, 0, stream>>>(stats, 128);
    int waves = NBATCH * (O / 16) * Nq;               // always divisible by 4
    k_gemm<O, CK, 0><<<waves / 4, 128, 0, stream>>>(wh, fk, fq, idx, gamma, beta,
                                                    mrs, stats, fout, Nq, Nk);
    float cnt = (float)((O / 4) * Nq * KNN_K);
    k_finalize<<<1, 64, 0, stream>>>(stats, mrs, cnt);
    k_gemm<O, CK, 1><<<waves / 4, 128, 0, stream>>>(wh, fk, fq, idx, gamma, beta,
                                                    mrs, stats, fout, Nq, Nk);
}

extern "C" void kernel_launch(void* const* d_in, const int* in_sizes, int n_in,
                              void* d_out, int out_size, void* d_ws, size_t ws_size,
                              hipStream_t stream) {
    (void)in_sizes; (void)n_in; (void)out_size; (void)ws_size;
    const float* x    = (const float*)d_in[0];
    const float* w_in = (const float*)d_in[3];
    const float* b_in = (const float*)d_in[4];
    const float* w1 = (const float*)d_in[5];
    const float* g1 = (const float*)d_in[6];
    const float* be1 = (const float*)d_in[7];
    const float* w2 = (const float*)d_in[8];
    const float* g2 = (const float*)d_in[9];
    const float* be2 = (const float*)d_in[10];
    const float* w3 = (const float*)d_in[11];
    const float* g3 = (const float*)d_in[12];
    const float* be3 = (const float*)d_in[13];
    const float* w4 = (const float*)d_in[14];
    const float* g4 = (const float*)d_in[15];
    const float* be4 = (const float*)d_in[16];

    // workspace carve (floats)
    float* ws = (float*)d_ws;
    size_t off = 0;
    auto alloc = [&](size_t nfl) { float* p = ws + off; off += nfl; return p; };
    float* coor0 = alloc((size_t)NBATCH * 3 * 2048);
    float* f0    = alloc((size_t)NBATCH * 2048 * 8);
    float* f1    = alloc((size_t)NBATCH * 2048 * 32);
    float* coor1 = alloc((size_t)NBATCH * 3 * 512);
    float* f1q   = alloc((size_t)NBATCH * 512 * 32);
    float* f2    = alloc((size_t)NBATCH * 512 * 64);
    float* f3    = alloc((size_t)NBATCH * 512 * 64);
    float* coor2 = alloc((size_t)NBATCH * 3 * 128);
    float* f3q   = alloc((size_t)NBATCH * 128 * 64);
    float* f4    = alloc((size_t)NBATCH * 128 * 128);
    float* stats = alloc(128);
    float* mrs   = alloc(128);
    _Float16* wh = (_Float16*)alloc(8192);                 // up to 128*128 halves
    int* idx     = (int*)alloc((size_t)NBATCH * 2048 * KNN_K);
    int* fidx    = (int*)alloc((size_t)NBATCH * 512);

    // input projection
    k_input<<<(NBATCH * 2048 + 255) / 256, 256, 0, stream>>>(x, w_in, b_in, coor0, f0);

    // stage 1: 2048 pts, C 8 -> 32
    launch_stage<32, 8>(w1, g1, be1, coor0, coor0, f0, f0, 2048, 2048, f1,
                        idx, wh, stats, mrs, stream);

    // FPS 2048 -> 512, gather
    k_fps<<<NBATCH, 256, 0, stream>>>(coor0, fidx, 2048, 512);
    k_gather<<<(NBATCH * 512 + 255) / 256, 256, 0, stream>>>(coor0, f1, fidx,
                                                             coor1, f1q, 2048, 32, 512);

    // stage 2: queries 512 vs keys 2048, C 32 -> 64
    launch_stage<64, 32>(w2, g2, be2, coor1, coor0, f1q, f1, 512, 2048, f2,
                         idx, wh, stats, mrs, stream);

    // stage 3: 512 pts, C 64 -> 64
    launch_stage<64, 64>(w3, g3, be3, coor1, coor1, f2, f2, 512, 512, f3,
                         idx, wh, stats, mrs, stream);

    // FPS 512 -> 128, gather
    k_fps<<<NBATCH, 256, 0, stream>>>(coor1, fidx, 512, 128);
    k_gather<<<(NBATCH * 128 + 255) / 256, 256, 0, stream>>>(coor1, f3, fidx,
                                                             coor2, f3q, 512, 64, 128);

    // stage 4: queries 128 vs keys 512, C 64 -> 128
    launch_stage<128, 64>(w4, g4, be4, coor2, coor1, f3q, f3, 128, 512, f4,
                          idx, wh, stats, mrs, stream);

    // outputs: (B,128,3) coords then (B,128,128) features
    int total = NBATCH * 128 * 3 + NBATCH * 128 * 128;
    k_out<<<(total + 255) / 256, 256, 0, stream>>>(coor2, f4, (float*)d_out);
}
